// EXPHORMER_2671469658761
// MI455X (gfx1250) — matrix operand order repaired and checked by reading, not compiled, not run on hardware
//
#include <hip/hip_runtime.h>
#include <hip/hip_bf16.h>
#include <cstdint>

// ---------------------------------------------------------------------------
// Exphormer graph-transformer forward for MI455X (gfx1250, wave32, WMMA).
// All dense GEMMs run on v_wmma_f32_16x16x32_bf16 (f32 accumulate).
// ---------------------------------------------------------------------------

typedef __attribute__((ext_vector_type(16))) __bf16 v16bf;
typedef __attribute__((ext_vector_type(8)))  float  v8f;

#define D_MODEL 256
#define QKV_W   768
#define FFN_W   1024
#define NHEAD   4
#define DH      64

#define TILE_M 128
#define TILE_N 64
#define TILE_K 32
#define AS_LD  40   // padded LDS row stride (bf16 elems): 80B, 16B-aligned, bank-spread
#define BS_LD  40

__device__ inline unsigned pack_bf16(float a, float b) {
    union { __bf16 h[2]; unsigned u; } t;
    t.h[0] = (__bf16)a;
    t.h[1] = (__bf16)b;
    return t.u;
}

// ---------------- WMMA fragment loaders (ISA 7.12.2 layouts) ----------------

// A fragment: 16x32 bf16 tile, row-major in LDS with row stride AS_LD.
// Lane layout: lanes 0-15 row M=lane, VGPR i<4 -> K=2i,2i+1 (+8 for lanes>=16),
// VGPR i>=4 -> K=16+2(i-4) (+8 for lanes>=16).  As uints this is exactly
// uint4 blocks [half] and [2+half] of the row -> two ds_load_b128.
__device__ inline v16bf load_a_frag(const __bf16* base, int lane) {
    int row  = lane & 15;
    int half = lane >> 4;
    union { v16bf v; uint4 q[2]; } f;
    const uint4* p = (const uint4*)(base + row * AS_LD);
    f.q[0] = p[half];
    f.q[1] = p[2 + half];
    return f.v;
}

// B fragment: 32x16 bf16 tile (K x N), stored COL-MAJOR in LDS as Bs[n][k]
// with row stride BS_LD.  Lane layout matches A: lanes 0-15 col N=lane hold
// K=0..7,16..23; lanes 16-31 hold K=8..15,24..31 -> two ds_load_b128.
__device__ inline v16bf load_b_frag(const __bf16* base, int lane) {
    int col  = lane & 15;
    int half = lane >> 4;
    union { v16bf v; uint4 q[2]; } f;
    const uint4* p = (const uint4*)(base + (size_t)col * BS_LD);
    f.q[0] = p[half];
    f.q[1] = p[2 + half];
    return f.v;
}

// ---------------- Generic GEMM: C = act(A @ W + bias) [+ residual] ----------
// A: M x K f32 (row-major), W: K x N f32 (row-major). K % 32 == 0, N % 64 == 0.
// 256 threads = 8 waves; waves tiled 4x2, each wave owns a 32x32 sub-tile.
template <bool GELU_ACT>
__global__ __launch_bounds__(256) void gemm_bias_kernel(
    const float* __restrict__ A, const float* __restrict__ W,
    const float* __restrict__ bias, float* __restrict__ C,
    const float* __restrict__ residual, int M, int K, int N) {
    __shared__ __bf16 As[TILE_M][AS_LD];   // row-major  [m][k]
    __shared__ __bf16 Bs[TILE_N][BS_LD];   // col-major  [n][k]

    const int tid   = threadIdx.x;
    const int lane  = tid & 31;
    const int wave  = tid >> 5;
    const int wr    = wave >> 1;   // 0..3
    const int wc    = wave & 1;    // 0..1
    const int mBase = blockIdx.y * TILE_M;
    const int nBase = blockIdx.x * TILE_N;

    v8f acc[2][2];
    const v8f vzero = {0.f, 0.f, 0.f, 0.f, 0.f, 0.f, 0.f, 0.f};
#pragma unroll
    for (int mi = 0; mi < 2; ++mi)
#pragma unroll
        for (int ni = 0; ni < 2; ++ni) acc[mi][ni] = vzero;

    for (int k0 = 0; k0 < K; k0 += TILE_K) {
        // ---- A tile: 128x32 f32 -> bf16 (1024 float4, 4 per thread) ----
#pragma unroll
        for (int it = 0; it < 4; ++it) {
            int f   = tid + it * 256;
            int row = f >> 3;            // 8 float4 per row
            int kc  = (f & 7) << 2;
            float4 v = make_float4(0.f, 0.f, 0.f, 0.f);
            int gRow = mBase + row;
            if (gRow < M) {
                const float* gp = A + (size_t)gRow * K + k0 + kc;
                v = *(const float4*)gp;
                if (k0 + TILE_K < K) __builtin_prefetch(gp + TILE_K, 0, 1);
            }
            unsigned* dp = (unsigned*)&As[row][kc];
            dp[0] = pack_bf16(v.x, v.y);
            dp[1] = pack_bf16(v.z, v.w);
        }
        // ---- B tile: 32x64 f32 -> bf16 col-major (1024 k-pairs, 4/thread) ---
        // consecutive lanes -> consecutive cols: coalesced b32 global loads;
        // each pair packs to one ds_store_b32 along k.
#pragma unroll
        for (int it = 0; it < 4; ++it) {
            int f   = tid + it * 256;
            int col = f & 63;
            int k   = (f >> 6) << 1;     // 0,2,...,30
            const float* gp = W + (size_t)(k0 + k) * N + nBase + col;
            float a = gp[0];
            float b = gp[N];
            *(unsigned*)&Bs[col][k] = pack_bf16(a, b);
        }
        __syncthreads();

        v16bf af[2], bfr[2];
        af[0]  = load_a_frag(&As[wr * 32][0], lane);
        af[1]  = load_a_frag(&As[wr * 32 + 16][0], lane);
        bfr[0] = load_b_frag(&Bs[wc * 32][0], lane);
        bfr[1] = load_b_frag(&Bs[wc * 32 + 16][0], lane);
#pragma unroll
        for (int mi = 0; mi < 2; ++mi)
#pragma unroll
            for (int ni = 0; ni < 2; ++ni)
                acc[mi][ni] = __builtin_amdgcn_wmma_f32_16x16x32_bf16(
                    false, af[mi], false, bfr[ni], (short)0, acc[mi][ni],
                    false, false);
        __syncthreads();
    }

    // ---- epilogue: C layout m = r + 8*(lane>=16), n = lane&15 ----
    const int halfsel = (lane >> 4) << 3;
    const int nlane   = lane & 15;
#pragma unroll
    for (int mi = 0; mi < 2; ++mi) {
#pragma unroll
        for (int ni = 0; ni < 2; ++ni) {
            int col = nBase + wc * 32 + ni * 16 + nlane;
#pragma unroll
            for (int r = 0; r < 8; ++r) {
                int m = mBase + wr * 32 + mi * 16 + halfsel + r;
                if (m < M) {
                    float v = acc[mi][ni][r] + bias[col];
                    if (GELU_ACT)
                        v = 0.5f * v * (1.0f + erff(v * 0.70710678118654752f));
                    if (residual) v += residual[(size_t)m * N + col];
                    C[(size_t)m * N + col] = v;
                }
            }
        }
    }
}

// ---------------- utility kernels -------------------------------------------

__global__ void zero_u32_kernel(unsigned* __restrict__ p, long long cnt) {
    long long i = (long long)blockIdx.x * blockDim.x + threadIdx.x;
    long long stride = (long long)gridDim.x * blockDim.x;
    for (; i < cnt; i += stride) p[i] = 0u;
}

__global__ void init_h_kernel(const float* __restrict__ x, float* __restrict__ h,
                              int n0) {
    long long i = (long long)blockIdx.x * blockDim.x + threadIdx.x;
    long long total = (long long)(n0 + 1) * D_MODEL;
    if (i >= total) return;
    long long row = i >> 8;  // D_MODEL == 256
    h[i] = (row < n0) ? x[i] : 0.f;
}

__global__ void copy_out_kernel(const float* __restrict__ h,
                                float* __restrict__ out, long long cnt) {
    long long i = (long long)blockIdx.x * blockDim.x + threadIdx.x;
    if (i < cnt) out[i] = h[i];
}

// local edges = [edge_index ; (i -> vn) ; (vn -> i)]
__global__ void build_loc_edges_kernel(const int* __restrict__ ei, int E, int n0,
                                       int* __restrict__ src, int* __restrict__ dst) {
    int i = blockIdx.x * blockDim.x + threadIdx.x;
    int total = E + 2 * n0;
    if (i >= total) return;
    if (i < E) {
        src[i] = ei[i];
        dst[i] = ei[E + i];
    } else if (i < E + n0) {
        src[i] = i - E;
        dst[i] = n0;
    } else {
        src[i] = n0;
        dst[i] = i - E - n0;
    }
}

// ---------------- attention (edge phase) -------------------------------------

__device__ inline unsigned encF(float f) {
    unsigned u = __float_as_uint(f);
    return (u & 0x80000000u) ? ~u : (u | 0x80000000u);
}
__device__ inline float decF(unsigned u) {
    u = (u & 0x80000000u) ? (u & 0x7FFFFFFFu) : ~u;
    return __uint_as_float(u);
}

// score[e,h] = dot(q[dst,h], k[src,h]) / 8 ; segment-max into mEnc[dst,h]
__global__ void attn_score_kernel(const float* __restrict__ qkv,
                                  const int* __restrict__ src,
                                  const int* __restrict__ dst, int nE,
                                  float* __restrict__ score,
                                  unsigned* __restrict__ mEnc) {
    int t = blockIdx.x * blockDim.x + threadIdx.x;
    if (t >= nE * NHEAD) return;
    int e = t >> 2, h = t & 3;
    int s = src[e], d = dst[e];
    const float4* q4 = (const float4*)(qkv + (size_t)d * QKV_W + h * DH);
    const float4* k4 = (const float4*)(qkv + (size_t)s * QKV_W + D_MODEL + h * DH);
    float a = 0.f;
#pragma unroll
    for (int i = 0; i < DH / 4; ++i) {
        float4 qa = q4[i], ka = k4[i];
        a += qa.x * ka.x + qa.y * ka.y + qa.z * ka.z + qa.w * ka.w;
    }
    a *= 0.125f;  // 1/sqrt(64)
    score[t] = a;
    atomicMax(&mEnc[d * NHEAD + h], encF(a));
}

// ex = exp(score - m[dst]); denom[dst] += ex (score buffer overwritten with ex)
__global__ void attn_expsum_kernel(float* __restrict__ score,
                                   const int* __restrict__ dst, int nE,
                                   const unsigned* __restrict__ mEnc,
                                   float* __restrict__ denom) {
    int t = blockIdx.x * blockDim.x + threadIdx.x;
    if (t >= nE * NHEAD) return;
    int e = t >> 2, h = t & 3;
    int d = dst[e];
    float m  = decF(mEnc[d * NHEAD + h]);
    float ex = expf(score[t] - m);
    score[t] = ex;
    atomicAdd(&denom[d * NHEAD + h], ex);
}

// out[dst] += (ex/denom[dst]) * v[src] ; one wave32 per edge, 8 floats/lane
__global__ void attn_aggregate_kernel(const float* __restrict__ qkv,
                                      const float* __restrict__ exv,
                                      const float* __restrict__ denom,
                                      const int* __restrict__ src,
                                      const int* __restrict__ dst, int nE,
                                      float* __restrict__ out) {
    int e    = (int)(((long long)blockIdx.x * blockDim.x + threadIdx.x) >> 5);
    int lane = threadIdx.x & 31;
    if (e >= nE) return;
    int s = src[e], d = dst[e];
    int col0 = lane * 8;          // 8 consecutive cols, fully within one head
    int h    = col0 >> 6;
    float alpha = exv[(size_t)e * NHEAD + h] /
                  (denom[(size_t)d * NHEAD + h] + 1e-16f);
    const float* v = qkv + (size_t)s * QKV_W + 2 * D_MODEL;
    float* o = out + (size_t)d * D_MODEL;
#pragma unroll
    for (int j = 0; j < 8; ++j) atomicAdd(&o[col0 + j], alpha * v[col0 + j]);
}

// hn = LN(h + t1 + t2) * g + b ; one 256-thread block per row
__global__ void ln_add_kernel(const float* __restrict__ h,
                              const float* __restrict__ t1,
                              const float* __restrict__ t2,
                              const float* __restrict__ g,
                              const float* __restrict__ b,
                              float* __restrict__ out, int n) {
    int row = blockIdx.x;
    if (row >= n) return;
    int tid = threadIdx.x;
    size_t idx = (size_t)row * D_MODEL + tid;
    float x = h[idx] + t1[idx] + t2[idx];
    __shared__ float red[D_MODEL];
    red[tid] = x;
    __syncthreads();
    for (int off = 128; off > 0; off >>= 1) {
        if (tid < off) red[tid] += red[tid + off];
        __syncthreads();
    }
    float mu = red[0] * (1.0f / D_MODEL);
    __syncthreads();
    float dxc = x - mu;
    red[tid] = dxc * dxc;
    __syncthreads();
    for (int off = 128; off > 0; off >>= 1) {
        if (tid < off) red[tid] += red[tid + off];
        __syncthreads();
    }
    float var = red[0] * (1.0f / D_MODEL);
    out[idx] = dxc * rsqrtf(var + 1e-5f) * g[tid] + b[tid];
}

// ---------------- host orchestration ----------------------------------------

extern "C" void kernel_launch(void* const* d_in, const int* in_sizes, int n_in,
                              void* d_out, int out_size, void* d_ws,
                              size_t ws_size, hipStream_t stream) {
    const float* x        = (const float*)d_in[0];
    const int*   eidx     = (const int*)d_in[1];
    const int*   xeidx    = (const int*)d_in[2];
    const float* Wqkv_loc = (const float*)d_in[3];
    const float* bqkv_loc = (const float*)d_in[4];
    const float* Wo_loc   = (const float*)d_in[5];
    const float* bo_loc   = (const float*)d_in[6];
    const float* Wqkv_exp = (const float*)d_in[7];
    const float* bqkv_exp = (const float*)d_in[8];
    const float* Wo_exp   = (const float*)d_in[9];
    const float* bo_exp   = (const float*)d_in[10];
    const float* ln_g     = (const float*)d_in[11];
    const float* ln_b     = (const float*)d_in[12];
    const float* W1       = (const float*)d_in[13];
    const float* b1       = (const float*)d_in[14];
    const float* W2       = (const float*)d_in[15];
    const float* b2       = (const float*)d_in[16];
    (void)n_in; (void)out_size; (void)ws_size;

    const int n0   = in_sizes[0] / D_MODEL;  // 20000
    const int E    = in_sizes[1] / 2;        // 320000
    const int EX   = in_sizes[2] / 2;        // 4*(n0+1)
    const int n    = n0 + 1;
    const int locE = E + 2 * n0;
    const int maxE = locE > EX ? locE : EX;

    char* p = (char*)d_ws;
    auto carve = [&](size_t bytes) -> char* {
        char* r = p;
        p += (bytes + 255) & ~(size_t)255;
        return r;
    };
    float*    h     = (float*)carve((size_t)n * D_MODEL * 4);
    float*    hn    = (float*)carve((size_t)n * D_MODEL * 4);
    float*    t1    = (float*)carve((size_t)n * D_MODEL * 4);
    float*    t2    = (float*)carve((size_t)n * D_MODEL * 4);
    float*    agg   = (float*)carve((size_t)n * D_MODEL * 4);
    float*    qkv   = (float*)carve((size_t)n * QKV_W * 4);
    float*    mid   = (float*)carve((size_t)n * FFN_W * 4);
    float*    sc    = (float*)carve((size_t)maxE * NHEAD * 4);
    unsigned* mEnc  = (unsigned*)carve((size_t)n * NHEAD * 4);
    float*    denom = (float*)carve((size_t)n * NHEAD * 4);
    int*      lsrc  = (int*)carve((size_t)locE * 4);
    int*      ldst  = (int*)carve((size_t)locE * 4);

    const dim3 blk(256);
    // init hidden state (append zero virtual-node row) + local edge lists
    {
        long long tot = (long long)n * D_MODEL;
        init_h_kernel<<<(unsigned)((tot + 255) / 256), blk, 0, stream>>>(x, h, n0);
        build_loc_edges_kernel<<<(locE + 255) / 256, blk, 0, stream>>>(eidx, E, n0,
                                                                       lsrc, ldst);
    }

    const int mBlocks = (n + TILE_M - 1) / TILE_M;

    auto run_attn = [&](const float* Wqkv, const float* bqkv, const float* Wo,
                        const float* bo, const int* src, const int* dst, int nE,
                        float* tout) {
        dim3 gq(QKV_W / TILE_N, mBlocks);
        gemm_bias_kernel<false><<<gq, blk, 0, stream>>>(h, Wqkv, bqkv, qkv,
                                                        nullptr, n, D_MODEL, QKV_W);
        long long nm = (long long)n * NHEAD;
        long long na = (long long)n * D_MODEL;
        zero_u32_kernel<<<(unsigned)((nm + 255) / 256), blk, 0, stream>>>(mEnc, nm);
        zero_u32_kernel<<<(unsigned)((nm + 255) / 256), blk, 0, stream>>>(
            (unsigned*)denom, nm);
        zero_u32_kernel<<<(unsigned)((na + 255) / 256), blk, 0, stream>>>(
            (unsigned*)agg, na);
        int nt = nE * NHEAD;
        attn_score_kernel<<<(nt + 255) / 256, blk, 0, stream>>>(qkv, src, dst, nE,
                                                                sc, mEnc);
        attn_expsum_kernel<<<(nt + 255) / 256, blk, 0, stream>>>(sc, dst, nE, mEnc,
                                                                 denom);
        attn_aggregate_kernel<<<(nE + 7) / 8, blk, 0, stream>>>(qkv, sc, denom, src,
                                                                dst, nE, agg);
        dim3 go(D_MODEL / TILE_N, mBlocks);
        gemm_bias_kernel<false><<<go, blk, 0, stream>>>(agg, Wo, bo, tout, nullptr,
                                                        n, D_MODEL, D_MODEL);
    };

    for (int l = 0; l < 3; ++l) {
        run_attn(Wqkv_loc + (size_t)l * D_MODEL * QKV_W,
                 bqkv_loc + (size_t)l * QKV_W,
                 Wo_loc + (size_t)l * D_MODEL * D_MODEL,
                 bo_loc + (size_t)l * D_MODEL, lsrc, ldst, locE, t1);
        run_attn(Wqkv_exp + (size_t)l * D_MODEL * QKV_W,
                 bqkv_exp + (size_t)l * QKV_W,
                 Wo_exp + (size_t)l * D_MODEL * D_MODEL,
                 bo_exp + (size_t)l * D_MODEL, xeidx, xeidx + EX, EX, t2);

        ln_add_kernel<<<n, blk, 0, stream>>>(h, t1, t2,
                                             ln_g + (size_t)l * D_MODEL,
                                             ln_b + (size_t)l * D_MODEL, hn, n);

        dim3 g1(FFN_W / TILE_N, mBlocks);
        gemm_bias_kernel<true><<<g1, blk, 0, stream>>>(
            hn, W1 + (size_t)l * D_MODEL * FFN_W, b1 + (size_t)l * FFN_W, mid,
            nullptr, n, D_MODEL, FFN_W);
        dim3 g2(D_MODEL / TILE_N, mBlocks);
        gemm_bias_kernel<false><<<g2, blk, 0, stream>>>(
            mid, W2 + (size_t)l * FFN_W * D_MODEL, b2 + (size_t)l * D_MODEL, h, hn,
            n, FFN_W, D_MODEL);
    }

    long long outCnt = (long long)n0 * D_MODEL;
    copy_out_kernel<<<(unsigned)((outCnt + 255) / 256), blk, 0, stream>>>(
        h, (float*)d_out, outCnt);
}
